// WOS_72842645340326
// MI455X (gfx1250) — matrix operand order repaired and checked by reading, not compiled, not run on hardware
//
#include <hip/hip_runtime.h>
#include <hip/hip_bf16.h>

typedef __attribute__((ext_vector_type(16))) _Float16 v16h;
typedef __attribute__((ext_vector_type(8)))  float    v8f;

#define KSZ   3
#define INC   16
#define OUTC  32
#define NB    4
#define HWD   130
#define HO    128
#define CKK   144   // INC*KSZ*KSZ ; 2*CKK = 288 = 9*32 -> zero-padding chunking

__device__ __forceinline__ float fast_tanh(float x) {
#if __has_builtin(__builtin_amdgcn_tanhf)
  return __builtin_amdgcn_tanhf(x);
#elif __has_builtin(__builtin_amdgcn_tanh_f32)
  return __builtin_amdgcn_tanh_f32(x);
#else
  return tanhf(x);
#endif
}

// Offset (in floats) of unfolded patch element k relative to x[b][0][r][col].
// k = c*9 + ki*3 + kj ; all call sites have compile-time k.
__device__ __forceinline__ constexpr int xoff(int k) {
  return (k / 9) * (HWD * HWD) + ((k % 9) / 3) * HWD + (k % 3);
}

// One 32-wide expanded-K chunk C (template param => everything folds at
// compile time; no runtime div/mod can survive).  Expanded K index
// KK = 32*C + (B/A lane-element map) in [0,288) covers a channel PAIR:
// channel = pairBase + KK/144, k = KK%144.
template <int C>
__device__ __forceinline__ void chunk_accum(const float* __restrict__ xb,
                                            const float* __restrict__ wrow,
                                            const float* __restrict__ mbase,
                                            int n, int h, v8f& acc) {
  // ---- patch values (pair-independent): B element e <-> KK = 32C+(e&7)+8h+16(e>>3)
  float    px[16];
  // ---- weight values (pair-independent): A element e <-> KA = 32C+(e&7)+8h+16(e>>3)
  _Float16 af[16];
#pragma unroll
  for (int e = 0; e < 16; ++e) {
    const int klo = (32 * C + (e & 7) + 16 * (e >> 3)) % CKK;      // h = 0
    const int khi = (32 * C + (e & 7) + 16 * (e >> 3) + 8) % CKK;  // h = 1
    const float vlo = xb[xoff(klo)];
    const float vhi = xb[xoff(khi)];
    px[e] = h ? vhi : vlo;

    const int kaL = 32 * C + (e & 7) + 16 * (e >> 3);
    const float wlo = wrow[kaL % CKK];             // h = 0
    const float whi = wrow[(kaL + 8) % CKK];       // h = 1 (same 16-block)
    af[e] = (_Float16)(h ? whi : wlo);
  }

  // ---- 8 channel pairs; each WMMA feeds D rows 2*j2 and 2*j2+1 ------------
  // Pair's two mask rows are CONTIGUOUS (stride 144 floats), so mrow[KK]
  // with KK in [0,288) addresses the whole pair segment directly.
  const float* __restrict__ mrow = mbase;
  for (int j2 = 0; j2 < 8; ++j2) {
    v16h bb;
#pragma unroll
    for (int e = 0; e < 16; ++e) {
      const int KKlo = 32 * C + (e & 7) + 16 * (e >> 3); // h = 0
      const int KKhi = KKlo + 8;                   // h = 1
      const float mv = h ? mrow[KKhi] : mrow[KKlo];
      bb[e] = (_Float16)fast_tanh(px[e] + mv);     // v_tanh_f32
    }

    // A mask: element belongs to pair row (KA >= 144); h-invariant and
    // dword-uniform => folds to 8 cndmasks on two wave masks.
    const bool m0 = (n == 2 * j2);
    const bool m1 = (n == 2 * j2 + 1);
    v16h aa;
#pragma unroll
    for (int e = 0; e < 16; ++e) {
      const bool rhi = (32 * C + (e & 7) + 16 * (e >> 3)) >= CKK; // constexpr
      aa[e] = (rhi ? m1 : m0) ? af[e] : (_Float16)0.f;
    }

    acc = __builtin_amdgcn_wmma_f32_16x16x32_f16(
        /*neg_a=*/false, aa, /*neg_b=*/false, bb,
        /*c_mod=*/(short)0, acc, /*reuse_a=*/false, /*reuse_b=*/false);

    mrow += 2 * CKK;
  }
}

__global__ __launch_bounds__(256)
void WOS_masked_conv_tanh_wmma(const float* __restrict__ x,
                               const float* __restrict__ w,
                               const float* __restrict__ msk,
                               const float* __restrict__ bias,
                               float* __restrict__ out) {
  const int lane = threadIdx.x & 31;
  const int wid  = blockIdx.x * 8 + (threadIdx.x >> 5);   // 0..8191, exact

  const int ctile = wid & 7;            // 16-column group within a row
  const int r     = (wid >> 3) & 127;   // output row
  const int ot    = (wid >> 10) & 1;    // o-tile (0: o=0..15, 1: o=16..31)
  const int b     = wid >> 11;          // batch

  const int n = lane & 15;              // B column / C column / A row of this lane
  const int h = lane >> 4;              // lane half (selects K sub-range)
  const int obase = ot * 16;
  const int col   = ctile * 16 + n;

  // x[b][0][r][col] ; all patch offsets are compile-time immediates
  const float* __restrict__ xb    = x + ((size_t)(b * INC) * HWD + r) * HWD + col;
  // weight row for this lane's A-matrix row m = lane%16
  const float* __restrict__ wrow  = w + (size_t)(obase + n) * CKK;
  const float* __restrict__ mbase = msk + (size_t)obase * CKK;

  v8f acc = {0.f, 0.f, 0.f, 0.f, 0.f, 0.f, 0.f, 0.f};

  // 9 chunks of 32 expanded-K each (2*144 = 288 = 9*32, zero padding).
  // Explicit template instantiations: guaranteed full unroll, all offsets
  // compile-time (round-2 regression left this loop runtime -> SALU div chains).
  chunk_accum<0>(xb, wrow, mbase, n, h, acc);
  chunk_accum<1>(xb, wrow, mbase, n, h, acc);
  chunk_accum<2>(xb, wrow, mbase, n, h, acc);
  chunk_accum<3>(xb, wrow, mbase, n, h, acc);
  chunk_accum<4>(xb, wrow, mbase, n, h, acc);
  chunk_accum<5>(xb, wrow, mbase, n, h, acc);
  chunk_accum<6>(xb, wrow, mbase, n, h, acc);
  chunk_accum<7>(xb, wrow, mbase, n, h, acc);
  chunk_accum<8>(xb, wrow, mbase, n, h, acc);

  // D layout: VGPR t holds row (t + 8*h), column n  ->  o = obase + t + 8*h
#pragma unroll
  for (int t = 0; t < 8; ++t) {
    const int o = obase + t + 8 * h;
    out[((size_t)(b * OUTC + o)) * (HO * HO) + (size_t)r * HO + col] =
        acc[t] - bias[o];
  }
}

extern "C" void kernel_launch(void* const* d_in, const int* in_sizes, int n_in,
                              void* d_out, int out_size, void* d_ws, size_t ws_size,
                              hipStream_t stream) {
  const float* x    = (const float*)d_in[0];  // (4,16,130,130)
  const float* wgt  = (const float*)d_in[1];  // (32,16,3,3)
  const float* msk  = (const float*)d_in[2];  // (32,16,3,3)
  const float* bias = (const float*)d_in[3];  // (32,1)
  float* out = (float*)d_out;                 // (4,32,128,128)

  // 8192 tiles (one wave each), 8 waves per 256-thread block
  WOS_masked_conv_tanh_wmma<<<dim3(1024), dim3(256), 0, stream>>>(
      x, wgt, msk, bias, out);
}
